// ParamAttention_283467841812
// MI455X (gfx1250) — compile-verified, matrix operand order checked by reading, not run on hardware
//
#include <hip/hip_runtime.h>

// ---------- types ----------
typedef __attribute__((ext_vector_type(16))) __bf16 v16bf;
typedef __attribute__((ext_vector_type(8)))  __bf16 v8bf;
typedef __attribute__((ext_vector_type(8)))  float  v8f;
typedef __attribute__((ext_vector_type(4)))  float  v4f;

union BF16x16 { v16bf v; v8bf h[2]; };

// Problem sizes
#define BB 16
#define QQ 1024
#define CC 2048
#define DD 512

__device__ __forceinline__ v8f wmma_bf16(v16bf a, v16bf b, v8f c) {
  return __builtin_amdgcn_wmma_f32_16x16x32_bf16(false, a, false, b, (short)0, c,
                                                 false, false);
}

// A-style fragment load from a row-major bf16 matrix (works for LDS or global).
// 16-bit A-matrix 16x32 layout: lane(0-15)=row, holds K= h*8..h*8+7 (v0-3) and
// 16+h*8.. (v4-7), where h = lane>>4.
__device__ __forceinline__ v16bf load_frag(const __bf16* base, int row0, int ld,
                                           int k0, int l16, int half) {
  const __bf16* p = base + (size_t)(row0 + l16) * ld + k0 + half * 8;
  BF16x16 r;
  r.h[0] = *(const v8bf*)(p);
  r.h[1] = *(const v8bf*)(p + 16);
  return r.v;
}

// CDNA5 async copy: 16 bytes global -> LDS per lane, tracked by ASYNCcnt.
// lds_off is a byte offset from the wave's LDS base (dynamic LDS starts at 0).
__device__ __forceinline__ void async_copy_b128(unsigned lds_off,
                                                const void* gaddr) {
  asm volatile("global_load_async_to_lds_b128 %0, %1, off"
               :: "v"(lds_off), "v"(gaddr)
               : "memory");
}
// Async loads complete in order per wave: waiting to <=N proves the oldest
// (outstanding - N) copies have landed.
#define ASYNC_WAIT(n) asm volatile("s_wait_asynccnt " #n ::: "memory")

// ---------- kernel 0: fp32 -> bf16 convert (weights) ----------
__global__ void cvt_f32_bf16_kernel(const float* __restrict__ in,
                                    __bf16* __restrict__ out, int n) {
  int i = (blockIdx.x * blockDim.x + threadIdx.x) * 4;
  if (i + 3 < n) {
    v4f x = *(const v4f*)(in + i);
    out[i + 0] = (__bf16)x.x;
    out[i + 1] = (__bf16)x.y;
    out[i + 2] = (__bf16)x.z;
    out[i + 3] = (__bf16)x.w;
  }
}

// ---------- kernel 1: projection  Y = X * W^T  (NT GEMM, bf16 out) ----------
// X: (Mtot, 512) fp32 row-major.  W: (512, 512) bf16 row-major.
// transpose==0: Y[m*512 + n]   (row-major, used for q and k)
// transpose==1: Y[b*512*MperB + n*MperB + mloc]  (vT, used for v)
// Block: 256 threads = 8 waves, tile 64 rows x 512 cols.
// Wave w: rows (w>>1)*16, cols (w&1)*256 .. +256 (16 n-tiles).
__global__ void __launch_bounds__(256, 1)
proj_kernel(const float* __restrict__ X, const __bf16* __restrict__ W,
            __bf16* __restrict__ Y, int MperB, int transpose) {
  extern __shared__ char smem[];
  __bf16* xs = (__bf16*)smem;  // [64][512] bf16 = 64KB

  const int tid  = threadIdx.x;
  const int row0 = blockIdx.x * 64;

  // Stage X tile to LDS with fp32->bf16 conversion (VALU path: cvt needed).
  {
    const float* src = X + (size_t)row0 * DD;
    for (int idx = tid * 4; idx < 64 * DD; idx += 256 * 4) {
      v4f x = *(const v4f*)(src + idx);
      xs[idx + 0] = (__bf16)x.x;
      xs[idx + 1] = (__bf16)x.y;
      xs[idx + 2] = (__bf16)x.z;
      xs[idx + 3] = (__bf16)x.w;
    }
  }
  __syncthreads();

  const int lane = tid & 31, l16 = lane & 15, half = lane >> 4;
  const int wv = tid >> 5;
  const int rg = wv >> 1;  // row group 0..3
  const int cg = wv & 1;   // col group 0..1

  v8f acc[16];
#pragma unroll
  for (int nt = 0; nt < 16; ++nt) acc[nt] = (v8f){0, 0, 0, 0, 0, 0, 0, 0};

  for (int k0 = 0; k0 < DD; k0 += 32) {
    v16bf a = load_frag(xs, rg * 16, DD, k0, l16, half);
#pragma unroll
    for (int nt = 0; nt < 16; ++nt) {
      // NT GEMM: B-operand fragment == A-style load of W rows (row = out col n)
      v16bf b = load_frag(W, cg * 256 + nt * 16, DD, k0, l16, half);
      acc[nt] = wmma_bf16(a, b, acc[nt]);
    }
  }

  // Epilogue. C-layout: VGPR i <-> row (i + half*8), lane l16 <-> col.
  const int mbase = row0 + rg * 16 + half * 8;
  if (!transpose) {
#pragma unroll
    for (int nt = 0; nt < 16; ++nt) {
      int n = cg * 256 + nt * 16 + l16;
#pragma unroll
      for (int i = 0; i < 8; ++i)
        Y[(size_t)(mbase + i) * DD + n] = (__bf16)acc[nt][i];
    }
  } else {
    const int bidx = row0 / MperB;
    const int mloc = (row0 % MperB) + rg * 16 + half * 8;
    __bf16* Yb = Y + (size_t)bidx * DD * MperB;
#pragma unroll
    for (int nt = 0; nt < 16; ++nt) {
      int n = cg * 256 + nt * 16 + l16;
#pragma unroll
      for (int i = 0; i < 8; ++i)
        Yb[(size_t)n * MperB + (mloc + i)] = (__bf16)acc[nt][i];
    }
  }
}

// ---------- kernel 2: flash attention (double-buffered async staging) ----------
// q: (B,1024,512) bf16, k: (B,2048,512) bf16, vT: (B,512,2048) bf16.
// Block = 128 threads (4 waves); block handles (b, 64 q-rows); wave = 16 rows,
// full D=512 output (32 accumulator tiles). C processed in chunks of 32.
// LDS byte map (dynamic region, offset 0), 200KB total:
//   qs    : [0,      65536)   64*512 bf16
//   ks[2] : [65536, 131072)   2 x 32*512 bf16
//   vs[2] : [131072,196608)   2 x 512*32 bf16
//   ps    : [196608,204800)   4 waves * 16*32 fp32
#define QS_OFF  0u
#define KS_OFF  65536u
#define VS_OFF  131072u
#define PS_OFF  196608u
#define KV_BUF  32768u

// Issue one chunk's async copies: 32 b128 per thread (16 for k, 16 for vT).
__device__ __forceinline__ void issue_chunk(const __bf16* __restrict__ kg,
                                            const __bf16* __restrict__ vtg,
                                            int b, int c0, int tid,
                                            unsigned buf) {
  const unsigned ksoff = KS_OFF + buf * KV_BUF;
  const unsigned vsoff = VS_OFF + buf * KV_BUF;
  const __bf16* ksrc = kg + ((size_t)b * CC + c0) * DD;
  for (int idx = tid * 8; idx < 32 * DD; idx += 128 * 8)
    async_copy_b128(ksoff + (unsigned)idx * 2u, ksrc + idx);
  const __bf16* vsrc = vtg + (size_t)b * DD * CC + c0;
  for (int e = tid * 8; e < DD * 32; e += 128 * 8) {
    int d = e >> 5, cc = e & 31;
    async_copy_b128(vsoff + (unsigned)e * 2u, vsrc + (size_t)d * CC + cc);
  }
}

__global__ void __launch_bounds__(128, 1)
attn_kernel(const __bf16* __restrict__ qg, const __bf16* __restrict__ kg,
            const __bf16* __restrict__ vtg,
            const unsigned char* __restrict__ mask, float* __restrict__ out) {
  extern __shared__ char smem[];
  __bf16* qs = (__bf16*)(smem + QS_OFF);
  float*  ps = (float*)(smem + PS_OFF);

  const int tid = threadIdx.x;
  const int b   = blockIdx.x >> 4;          // Q/64 = 16 tiles per batch
  const int q0  = (blockIdx.x & 15) * 64;
  const int lane = tid & 31, l16 = lane & 15, half = lane >> 4;
  const int wv = tid >> 5;  // 0..3

  // Stage q tile (64 rows) once via async DMA, then prefetch chunk 0.
  {
    const __bf16* src = qg + ((size_t)b * QQ + q0) * DD;
    for (int idx = tid * 8; idx < 64 * DD; idx += 128 * 8)
      async_copy_b128(QS_OFF + (unsigned)idx * 2u, src + idx);
  }
  issue_chunk(kg, vtg, b, 0, tid, 0u);

  float mI[8], lI[8];
#pragma unroll
  for (int i = 0; i < 8; ++i) { mI[i] = -INFINITY; lI[i] = 0.f; }
  v8f acc[32];
#pragma unroll
  for (int nt = 0; nt < 32; ++nt) acc[nt] = (v8f){0, 0, 0, 0, 0, 0, 0, 0};

  float* myps = ps + wv * (16 * 32);
  // fold 1/sqrt(D) and log2(e) so softmax runs in exp2 domain (v_exp_f32)
  const float sc = 0.044194173824159216f * 1.4426950408889634f;

  for (int ci = 0; ci < CC / 32; ++ci) {
    const unsigned buf = (unsigned)(ci & 1);
    __bf16* ks = (__bf16*)(smem + KS_OFF + buf * KV_BUF);
    __bf16* vs = (__bf16*)(smem + VS_OFF + buf * KV_BUF);

    if (ci + 1 < CC / 32) {
      // Prev iteration's trailing barrier proved buf^1 has no readers left.
      issue_chunk(kg, vtg, b, (ci + 1) * 32, tid, buf ^ 1u);
      ASYNC_WAIT(0x20);  // 32 outstanding (next chunk) => chunk ci landed
    } else {
      ASYNC_WAIT(0x0);
    }
    __syncthreads();  // everyone's chunk-ci copies are visible in LDS

    // s(16x32) = q_tile . k_chunk^T  (NT: both operands A-style loads)
    v8f s0 = (v8f){0, 0, 0, 0, 0, 0, 0, 0};
    v8f s1 = (v8f){0, 0, 0, 0, 0, 0, 0, 0};
#pragma unroll
    for (int k0 = 0; k0 < DD; k0 += 32) {
      v16bf a  = load_frag(qs, wv * 16, DD, k0, l16, half);
      v16bf b0 = load_frag(ks, 0, DD, k0, l16, half);
      v16bf b1 = load_frag(ks, 16, DD, k0, l16, half);
      s0 = wmma_bf16(a, b0, s0);
      s1 = wmma_bf16(a, b1, s1);
    }

    // Online softmax (base-2). C-layout: VGPR i = row i/i+8, lanes = cols;
    // xor-shuffles 1/2/4/8 reduce across each 16-lane half independently.
    float corr[8];
#pragma unroll
    for (int i = 0; i < 8; ++i) {
      s0[i] *= sc; s1[i] *= sc;
      float t = fmaxf(s0[i], s1[i]);
      t = fmaxf(t, __shfl_xor(t, 1, 32));
      t = fmaxf(t, __shfl_xor(t, 2, 32));
      t = fmaxf(t, __shfl_xor(t, 4, 32));
      t = fmaxf(t, __shfl_xor(t, 8, 32));
      float mNew = fmaxf(mI[i], t);
      corr[i] = exp2f(mI[i] - mNew);
      float p0 = exp2f(s0[i] - mNew);
      float p1 = exp2f(s1[i] - mNew);
      s0[i] = p0; s1[i] = p1;
      float rs = p0 + p1;
      rs += __shfl_xor(rs, 1, 32);
      rs += __shfl_xor(rs, 2, 32);
      rs += __shfl_xor(rs, 4, 32);
      rs += __shfl_xor(rs, 8, 32);
      lI[i] = lI[i] * corr[i] + rs;
      mI[i] = mNew;
    }
#pragma unroll
    for (int nt = 0; nt < 32; ++nt)
#pragma unroll
      for (int i = 0; i < 8; ++i) acc[nt][i] *= corr[i];

    // Re-layout p: C-format -> A-format through wave-private LDS scratch.
#pragma unroll
    for (int i = 0; i < 8; ++i) {
      myps[(i + half * 8) * 32 + l16]      = s0[i];
      myps[(i + half * 8) * 32 + 16 + l16] = s1[i];
    }
    asm volatile("s_wait_dscnt 0x0" ::: "memory");
    v16bf pf;
    {
      const float* pr = myps + l16 * 32 + half * 8;
      BF16x16 r;
#pragma unroll
      for (int j = 0; j < 8; ++j) {
        r.h[0][j] = (__bf16)pr[j];        // K = half*8 + j
        r.h[1][j] = (__bf16)pr[16 + j];   // K = 16 + half*8 + j
      }
      pf = r.v;
    }

    // out += p . v_chunk ; B-operand from vs rows (d-major => A-style load)
#pragma unroll
    for (int nt = 0; nt < 32; ++nt) {
      v16bf bf = load_frag(vs, nt * 16, 32, 0, l16, half);
      acc[nt] = wmma_bf16(pf, bf, acc[nt]);
    }

    __syncthreads();  // all readers of buf done; next iter may overwrite it
  }

  // Epilogue: normalize by l, apply mask, store fp32.
  const int rowb = q0 + wv * 16 + half * 8;
  float fm[8];
#pragma unroll
  for (int i = 0; i < 8; ++i) {
    float inv = 1.0f / lI[i];
    fm[i] = mask[(size_t)b * QQ + rowb + i] ? inv : 0.0f;
  }
  float* ob = out + (size_t)b * QQ * DD;
#pragma unroll
  for (int nt = 0; nt < 32; ++nt) {
#pragma unroll
    for (int i = 0; i < 8; ++i)
      ob[(size_t)(rowb + i) * DD + nt * 16 + l16] = acc[nt][i] * fm[i];
  }
}

// ---------- host ----------
extern "C" void kernel_launch(void* const* d_in, const int* in_sizes, int n_in,
                              void* d_out, int out_size, void* d_ws,
                              size_t ws_size, hipStream_t stream) {
  const float* x_k = (const float*)d_in[0];
  const float* x_q = (const float*)d_in[1];
  const unsigned char* x_mask = (const unsigned char*)d_in[2];
  const float* Wq = (const float*)d_in[3];
  const float* Wk = (const float*)d_in[4];
  const float* Wv = (const float*)d_in[5];
  float* out = (float*)d_out;

  char* ws = (char*)d_ws;
  const size_t WB = (size_t)DD * DD * 2;  // 512KB per weight (bf16)
  __bf16* Wqb = (__bf16*)(ws);
  __bf16* Wkb = (__bf16*)(ws + WB);
  __bf16* Wvb = (__bf16*)(ws + 2 * WB);
  __bf16* qb  = (__bf16*)(ws + 3 * WB);
  __bf16* kb  = (__bf16*)(ws + 3 * WB + (size_t)BB * QQ * DD * 2);
  __bf16* vtb = (__bf16*)(ws + 3 * WB + (size_t)BB * QQ * DD * 2 +
                          (size_t)BB * CC * DD * 2);

  // 0) weights -> bf16
  {
    int n = DD * DD;
    cvt_f32_bf16_kernel<<<n / (256 * 4), 256, 0, stream>>>(Wq, Wqb, n);
    cvt_f32_bf16_kernel<<<n / (256 * 4), 256, 0, stream>>>(Wk, Wkb, n);
    cvt_f32_bf16_kernel<<<n / (256 * 4), 256, 0, stream>>>(Wv, Wvb, n);
  }

  // 1) projections
  const int projSmem = 64 * DD * 2;  // 64KB
  hipFuncSetAttribute((const void*)proj_kernel,
                      hipFuncAttributeMaxDynamicSharedMemorySize, projSmem);
  proj_kernel<<<(BB * QQ) / 64, 256, projSmem, stream>>>(x_q, Wqb, qb, QQ, 0);
  proj_kernel<<<(BB * CC) / 64, 256, projSmem, stream>>>(x_k, Wkb, kb, CC, 0);
  proj_kernel<<<(BB * CC) / 64, 256, projSmem, stream>>>(x_k, Wvb, vtb, CC, 1);

  // 2) flash attention
  const int attSmem = PS_OFF + 4 * 16 * 32 * 4;  // 200KB
  hipFuncSetAttribute((const void*)attn_kernel,
                      hipFuncAttributeMaxDynamicSharedMemorySize, attSmem);
  attn_kernel<<<BB * (QQ / 64), 128, attSmem, stream>>>(qb, kb, vtb, x_mask,
                                                        out);
}
